// pBLSTM_14044543058555
// MI455X (gfx1250) — matrix operand order repaired and checked by reading, not compiled, not run on hardware
//
#include <hip/hip_runtime.h>

// pBLSTM: B=32, T=4096, D_IN=256, H=256 -> xr (32,2048,512), out (32,2048,512) + lens2
#define T2    2048
#define NB    32
#define G4    1024           // 4*H
#define DK    512            // 2*D_IN
#define HPAD  264            // 256 + 8 bf16 pad -> LDS bank stagger
#define OUT_ELEMS 33554432ull // 32*2048*512

typedef __attribute__((ext_vector_type(16))) __bf16        v16bf;
typedef __attribute__((ext_vector_type(8)))  float         v8f;
typedef __attribute__((ext_vector_type(4)))  float         f32x4;
typedef __attribute__((ext_vector_type(4)))  unsigned int  u32x4;

__device__ __forceinline__ __bf16 f2bf(float f) { return (__bf16)f; }  // v_cvt, RNE
__device__ __forceinline__ float sigm(float x)  { return 1.f / (1.f + __expf(-x)); }
__device__ __forceinline__ float ftanh(float x) { return 2.f / (1.f + __expf(-2.f * x)) - 1.f; }

// ---------------- prep: f32->bf16 weights, fused bias, lens2 -----------------
__global__ void pblstm_prep(const float* __restrict__ wihf, const float* __restrict__ whhf,
                            const float* __restrict__ bihf, const float* __restrict__ bhhf,
                            const float* __restrict__ wihb, const float* __restrict__ whhb,
                            const float* __restrict__ bihb, const float* __restrict__ bhhb,
                            const int* __restrict__ xlens,
                            __bf16* __restrict__ wihf_bf, __bf16* __restrict__ wihb_bf,
                            __bf16* __restrict__ whhf_bf, __bf16* __restrict__ whhb_bf,
                            float* __restrict__ biasf, float* __restrict__ biasb,
                            int* __restrict__ lens2, int* __restrict__ lens_out) {
  int i = blockIdx.x * blockDim.x + threadIdx.x;
  if (i < G4 * DK) { wihf_bf[i] = f2bf(wihf[i]); wihb_bf[i] = f2bf(wihb[i]); }
  if (i < G4 * 256) { whhf_bf[i] = f2bf(whhf[i]); whhb_bf[i] = f2bf(whhb[i]); }
  if (i < G4) { biasf[i] = bihf[i] + bhhf[i]; biasb[i] = bihb[i] + bhhb[i]; }
  if (i < NB) { int l2 = xlens[i] >> 1; lens2[i] = l2; lens_out[i] = l2; }
}

// ---------------- x -> bf16 (vectorized, one pass) ---------------------------
__global__ void pblstm_xconv(const float* __restrict__ x, __bf16* __restrict__ xbf, size_t n8) {
  size_t i = (size_t)blockIdx.x * blockDim.x + threadIdx.x;
  if (i < n8) {
    f32x4 a = *reinterpret_cast<const f32x4*>(x + i * 8);
    f32x4 b = *reinterpret_cast<const f32x4*>(x + i * 8 + 4);
    union { __bf16 h[8]; u32x4 q; } u;
#pragma unroll
    for (int j = 0; j < 4; ++j) { u.h[j] = f2bf(a[j]); u.h[4 + j] = f2bf(b[j]); }
    *reinterpret_cast<u32x4*>(xbf + i * 8) = u.q;
  }
}

__global__ void pblstm_zero(float* __restrict__ out, size_t n) {
  size_t i = (size_t)blockIdx.x * blockDim.x + threadIdx.x;
  if (i < n) out[i] = 0.f;
}

// -------- phase 1: gx[dir] = xr(_rev) @ w_ih^T   (WMMA bf16, f32 acc) --------
// wave computes a 16(M) x 64(N) tile over K=512; dir=1 gathers reversed rows.
__global__ void __launch_bounds__(256)
pblstm_gemm(const __bf16* __restrict__ xbf,
            const __bf16* __restrict__ wihf, const __bf16* __restrict__ wihb,
            const int* __restrict__ lens2,
            float* __restrict__ gxf, float* __restrict__ gxb) {
  const int dir = blockIdx.z;
  const __bf16* wih = dir ? wihb : wihf;
  float* gx = dir ? gxb : gxf;

  const int tid = threadIdx.x;
  const int wave = tid >> 5, lane = tid & 31;
  const int lm = lane & 15, lh = lane >> 4;

  const int rowtile = blockIdx.x * 8 + wave;   // 0..4095
  const int m0 = rowtile * 16;
  const int g0 = blockIdx.y * 64;

  // per-lane A source row (batch b, step tt), reversed+clamped for bwd dir
  const int m  = m0 + lm;
  const int b  = m >> 11;
  const int tt = m & (T2 - 1);
  int srow = tt;
  if (dir) { int L = lens2[b]; srow = L - 1 - tt; if (srow < 0) srow = 0; }
  const __bf16* xrow = xbf + ((size_t)b * T2 + srow) * DK;

  v8f acc[4];
#pragma unroll
  for (int n = 0; n < 4; ++n)
#pragma unroll
    for (int r = 0; r < 8; ++r) acc[n][r] = 0.f;

#pragma unroll
  for (int kt = 0; kt < 16; ++kt) {
    // A fragment: K = {kb..kb+15}, kb = lh*16
    union { v16bf v; u32x4 q[2]; } au;
    const __bf16* ap = xrow + kt * 32 + lh * 16;
    au.q[0] = *reinterpret_cast<const u32x4*>(ap);
    au.q[1] = *reinterpret_cast<const u32x4*>(ap + 8);
#pragma unroll
    for (int n = 0; n < 4; ++n) {
      const int g = g0 + n * 16 + lm;
      v16bf bv = *reinterpret_cast<const v16bf*>(wih + (size_t)g * DK + kt * 32 + lh * 16);
      acc[n] = __builtin_amdgcn_wmma_f32_16x16x32_bf16(false, au.v, false, bv,
                                                       (short)0, acc[n], false, false);
    }
  }
#pragma unroll
  for (int n = 0; n < 4; ++n) {
    const int g = g0 + n * 16 + lm;
#pragma unroll
    for (int r = 0; r < 8; ++r)
      gx[(size_t)(m0 + r + 8 * lh) * G4 + g] = acc[n][r];
  }
}

// -------- phase 2: sequential scan. 4 WGs = {fwd,bwd} x {batch half}. --------
// Wave w owns hidden cols [16w,16w+16): gate tiles N = {0,256,512,768}+16w,
// so i/f/g/o for its columns stay in-register. w_hh split: gates i,f (g<512)
// staged in 270KB of LDS; gates g,o persistent in 128 VGPRs -> no spill risk.
__global__ void __launch_bounds__(512) __attribute__((amdgpu_waves_per_eu(1)))
pblstm_rec(const float* __restrict__ gxf, const float* __restrict__ gxb,
           const __bf16* __restrict__ whhf, const __bf16* __restrict__ whhb,
           const float* __restrict__ biasf, const float* __restrict__ biasb,
           const int* __restrict__ lens2, float* __restrict__ out) {
  const int dir = blockIdx.x >> 1;
  const int b0  = (blockIdx.x & 1) * 16;
  const float*  gx   = dir ? gxb : gxf;
  const __bf16* whh  = dir ? whhb : whhf;
  const float*  bias = dir ? biasb : biasf;

  const int tid = threadIdx.x;
  const int wave = tid >> 5, lane = tid & 31;
  const int lm = lane & 15, lh = lane >> 4;

  extern __shared__ __bf16 smem_bf[];
  __bf16* Bsh = smem_bf;                 // [512][HPAD] : w_hh rows g in [0,512)
  __bf16* hsh = smem_bf + 512 * HPAD;    // [16][HPAD]  : h state (bf16)

  // stage w_hh gates i,f into LDS (16B chunks, bank-staggered rows)
  for (int c = tid; c < 512 * 32; c += 512) {
    const int row = c >> 5, col = (c & 31) * 8;
    *reinterpret_cast<u32x4*>(&Bsh[row * HPAD + col]) =
        *reinterpret_cast<const u32x4*>(whh + (size_t)row * 256 + col);
  }

  // persistent B fragments for gate tiles 2 (g) and 3 (o): 16 x v16bf = 128 VGPRs
  v16bf Bf[16];
#pragma unroll
  for (int t4 = 2; t4 < 4; ++t4)
#pragma unroll
    for (int kt = 0; kt < 8; ++kt) {
      const int g = t4 * 256 + wave * 16 + lm;
      Bf[(t4 - 2) * 8 + kt] =
          *reinterpret_cast<const v16bf*>(whh + (size_t)g * 256 + kt * 32 + lh * 16);
    }

  float cst[8], hreg[8], bv[4];
  int lenr[8];
  const float* gxp[8];
#pragma unroll
  for (int r = 0; r < 8; ++r) {
    cst[r] = 0.f; hreg[r] = 0.f;
    const int brow = b0 + r + 8 * lh;
    lenr[r] = lens2[brow];
    gxp[r] = gx + (size_t)brow * T2 * G4 + wave * 16 + lm;
  }
#pragma unroll
  for (int t4 = 0; t4 < 4; ++t4) bv[t4] = bias[t4 * 256 + wave * 16 + lm];

  for (int i = tid; i < 16 * HPAD; i += 512) hsh[i] = f2bf(0.f);
  __syncthreads();

  for (int t = 0; t < T2; ++t) {
    // C init: pre-activations from phase 1 (+ fused bias)
    v8f acc[4];
#pragma unroll
    for (int t4 = 0; t4 < 4; ++t4)
#pragma unroll
      for (int r = 0; r < 8; ++r)
        acc[t4][r] = gxp[r][(size_t)t * G4 + t4 * 256] + bv[t4];

    // recurrent GEMM: acc += h @ w_hh^T  (B: tiles 0,1 from LDS; 2,3 from regs)
#pragma unroll
    for (int kt = 0; kt < 8; ++kt) {
      union { v16bf v; u32x4 q[2]; } au;
      const __bf16* ap = &hsh[lm * HPAD + kt * 32 + lh * 16];
      au.q[0] = *reinterpret_cast<const u32x4*>(ap);
      au.q[1] = *reinterpret_cast<const u32x4*>(ap + 8);
#pragma unroll
      for (int t4 = 0; t4 < 2; ++t4) {
        union { v16bf v; u32x4 q[2]; } bu;
        const __bf16* bp = &Bsh[(t4 * 256 + wave * 16 + lm) * HPAD + kt * 32 + lh * 16];
        bu.q[0] = *reinterpret_cast<const u32x4*>(bp);
        bu.q[1] = *reinterpret_cast<const u32x4*>(bp + 8);
        acc[t4] = __builtin_amdgcn_wmma_f32_16x16x32_bf16(false, au.v, false, bu.v,
                                                          (short)0, acc[t4], false, false);
      }
#pragma unroll
      for (int t4 = 2; t4 < 4; ++t4)
        acc[t4] = __builtin_amdgcn_wmma_f32_16x16x32_bf16(false, au.v, false,
                                                          Bf[(t4 - 2) * 8 + kt],
                                                          (short)0, acc[t4], false, false);
    }
    __syncthreads();   // all reads of hsh done

    // cell update (register-local: wave owns i/f/g/o for its columns)
#pragma unroll
    for (int r = 0; r < 8; ++r) {
      const float iv = acc[0][r], fv = acc[1][r], gv = acc[2][r], ov = acc[3][r];
      const float cn = sigm(fv) * cst[r] + sigm(iv) * ftanh(gv);
      const float hn = sigm(ov) * ftanh(cn);
      const bool mk = t < lenr[r];
      cst[r] = mk ? cn : cst[r];
      const float hw = mk ? hn : hreg[r];
      hreg[r] = hw;
      hsh[(r + 8 * lh) * HPAD + wave * 16 + lm] = f2bf(hw);
      if (mk) {
        const int orow = dir ? (lenr[r] - 1 - t) : t;
        out[((size_t)(b0 + r + 8 * lh) * T2 + orow) * DK + dir * 256 + wave * 16 + lm] = hn;
      }
    }
    __syncthreads();   // hsh writes visible for next step
  }
}

extern "C" void kernel_launch(void* const* d_in, const int* in_sizes, int n_in,
                              void* d_out, int out_size, void* d_ws, size_t ws_size,
                              hipStream_t stream) {
  const float* x     = (const float*)d_in[0];
  const int*   xlens = (const int*)d_in[1];
  const float* wihf  = (const float*)d_in[2];
  const float* whhf  = (const float*)d_in[3];
  const float* bihf  = (const float*)d_in[4];
  const float* bhhf  = (const float*)d_in[5];
  const float* wihb  = (const float*)d_in[6];
  const float* whhb  = (const float*)d_in[7];
  const float* bihb  = (const float*)d_in[8];
  const float* bhhb  = (const float*)d_in[9];
  float* out = (float*)d_out;
  int* lens_out = (int*)d_out + OUT_ELEMS;   // tuple tail: lens2 (int32)

  char* ws = (char*)d_ws;
  size_t off = 0;
  auto take = [&](size_t bytes) { char* p = ws + off; off = (off + bytes + 255) & ~(size_t)255; return p; };
  float*  gxf     = (float*)take((size_t)NB * T2 * G4 * 4);   // 256 MB
  float*  gxb     = (float*)take((size_t)NB * T2 * G4 * 4);   // 256 MB
  __bf16* xbf     = (__bf16*)take(OUT_ELEMS * 2);             // 64 MB
  __bf16* wihf_bf = (__bf16*)take((size_t)G4 * DK * 2);
  __bf16* wihb_bf = (__bf16*)take((size_t)G4 * DK * 2);
  __bf16* whhf_bf = (__bf16*)take((size_t)G4 * 256 * 2);
  __bf16* whhb_bf = (__bf16*)take((size_t)G4 * 256 * 2);
  float*  biasf   = (float*)take(G4 * 4);
  float*  biasb   = (float*)take(G4 * 4);
  int*    lens2   = (int*)take(NB * 4);

  pblstm_prep<<<(G4 * DK + 255) / 256, 256, 0, stream>>>(
      wihf, whhf, bihf, bhhf, wihb, whhb, bihb, bhhb, xlens,
      wihf_bf, wihb_bf, whhf_bf, whhb_bf, biasf, biasb, lens2, lens_out);

  const size_t n8 = OUT_ELEMS / 8;
  pblstm_xconv<<<(unsigned)((n8 + 255) / 256), 256, 0, stream>>>(x, xbf, n8);

  pblstm_zero<<<(unsigned)((OUT_ELEMS + 255) / 256), 256, 0, stream>>>(out, OUT_ELEMS);

  pblstm_gemm<<<dim3(512, 16, 2), 256, 0, stream>>>(xbf, wihf_bf, wihb_bf, lens2, gxf, gxb);

  const int rec_lds = (512 + 16) * HPAD * 2;   // 278,784 B < 320 KB WGP LDS
  (void)hipFuncSetAttribute((const void*)pblstm_rec,
                            hipFuncAttributeMaxDynamicSharedMemorySize, rec_lds);
  pblstm_rec<<<4, 512, rec_lds, stream>>>(gxf, gxb, whhf_bf, whhb_bf, biasf, biasb, lens2, out);
}
